// IdentityAwaredCalibModule_v2_51556787421405
// MI455X (gfx1250) — matrix operand order repaired and checked by reading, not compiled, not run on hardware
//
#include <hip/hip_runtime.h>
#include <hip/hip_bf16.h>

#define N_SEQ 256
#define BATCH 1024
#define DIM   128
#define HID   64
#define TSTEPS 24
#define ODIM  8

typedef __attribute__((ext_vector_type(16))) __bf16 v16bf;
typedef __attribute__((ext_vector_type(8)))  float  v8f;

// ---------------------------------------------------------------- helpers
__device__ __forceinline__ v8f wmma_bf16(v16bf a, v16bf b, v8f c) {
  // D = A(16x32 bf16) * B(32x16 bf16) + C(16x16 f32)
  return __builtin_amdgcn_wmma_f32_16x16x32_bf16(false, a, false, b, (short)0, c,
                                                 false, false);
}

// A fragment (16x32, bf16) built on the fly from an f32 row-major tile.
// Lane L<16: row L, K in {0..15}; lane L>=16: row L-16, K in {16..31}.
__device__ __forceinline__ v16bf load_a_frag_f32(const float* __restrict__ p, int ld) {
  int lane = threadIdx.x & 31;
  const float* r = p + (size_t)(lane & 15) * ld + ((lane >> 4) << 4);
  v16bf a;
#pragma unroll
  for (int j = 0; j < 8; ++j) a[j] = (__bf16)r[j];
#pragma unroll
  for (int j = 0; j < 8; ++j) a[8 + j] = (__bf16)r[8 + j];
  return a;
}

// B fragment (32x16, bf16) from weight W[h_out, k] row-major (col n of B = row n of W).
// Lane L: col L&15, K = (L>>4)*16 + j  (16 contiguous bf16 = 32B per lane).
__device__ __forceinline__ v16bf load_b_frag(const __bf16* __restrict__ w, int ld) {
  int lane = threadIdx.x & 31;
  const __bf16* r = w + (size_t)(lane & 15) * ld + ((lane >> 4) << 4);
  v16bf b;
#pragma unroll
  for (int j = 0; j < 16; ++j) b[j] = r[j];
  return b;
}

__device__ __forceinline__ float fast_exp(float x) {
  return __builtin_amdgcn_exp2f(x * 1.4426950408889634f);
}
__device__ __forceinline__ float fast_tanh(float x) {
#if __has_builtin(__builtin_amdgcn_tanhf)
  return __builtin_amdgcn_tanhf(x);          // v_tanh_f32 (CDNA5 transcendental)
#else
  return 1.0f - 2.0f * __builtin_amdgcn_rcpf(fast_exp(2.0f * x) + 1.0f);
#endif
}
__device__ __forceinline__ float fast_sig(float x) {
  return __builtin_amdgcn_rcpf(1.0f + fast_exp(-x));
}

// ---------------------------------------------------------------- kernels
__global__ void k_cvt(const float* __restrict__ s, __bf16* __restrict__ d, int n) {
  int i = blockIdx.x * blockDim.x + threadIdx.x;
  if (i < n) d[i] = (__bf16)s[i];
}

// Fused dual projection: out0/1[b,n,h] = x[n,b,:] . W0/1[h,:] + bias0/1[h]
// x:[N,B,D] read ONCE for both weight matrices. grid (B/64, N), 128 thr (4 waves x 16 rows).
__global__ void k_gemm_xw2(const float* __restrict__ x,
                           const __bf16* __restrict__ W0, const float* __restrict__ bias0,
                           float* __restrict__ out0,
                           const __bf16* __restrict__ W1, const float* __restrict__ bias1,
                           float* __restrict__ out1) {
  int w = threadIdx.x >> 5, lane = threadIdx.x & 31;
  int b0 = blockIdx.x * 64 + w * 16;
  int n  = blockIdx.y;
  const float* A = x + ((size_t)n * BATCH + b0) * DIM;
  v8f acc0[4] = {}, acc1[4] = {};
#pragma unroll
  for (int kk = 0; kk < 4; ++kk) {
    v16bf a = load_a_frag_f32(A + kk * 32, DIM);
#pragma unroll
    for (int nt = 0; nt < 4; ++nt) {
      acc0[nt] = wmma_bf16(a, load_b_frag(W0 + (size_t)(nt * 16) * DIM + kk * 32, DIM), acc0[nt]);
      acc1[nt] = wmma_bf16(a, load_b_frag(W1 + (size_t)(nt * 16) * DIM + kk * 32, DIM), acc1[nt]);
    }
  }
  int col = lane & 15, m0 = (lane >> 4) << 3;
#pragma unroll
  for (int nt = 0; nt < 4; ++nt) {
    int hc = nt * 16 + col;
    float bi0 = bias0[hc], bi1 = bias1[hc];
#pragma unroll
    for (int r = 0; r < 8; ++r) {
      size_t o = ((size_t)(b0 + m0 + r) * N_SEQ + n) * HID + hc;
      out0[o] = acc0[nt][r] + bi0;
      out1[o] = acc1[nt][r] + bi1;
    }
  }
}

// E[row,h'] = Y[row,:] . Wfc[h',:]   (Y:[R,64] f32 flat, R = B*N)  grid R/64, 128 thr.
__global__ void k_gemm_enc(const float* __restrict__ Y, const __bf16* __restrict__ Wb,
                           float* __restrict__ E) {
  int w = threadIdx.x >> 5, lane = threadIdx.x & 31;
  int r0 = blockIdx.x * 64 + w * 16;
  const float* A = Y + (size_t)r0 * HID;
  v8f acc[4] = {};
#pragma unroll
  for (int kk = 0; kk < 2; ++kk) {
    v16bf a = load_a_frag_f32(A + kk * 32, HID);
#pragma unroll
    for (int nt = 0; nt < 4; ++nt)
      acc[nt] = wmma_bf16(a, load_b_frag(Wb + (size_t)(nt * 16) * HID + kk * 32, HID), acc[nt]);
  }
  int col = lane & 15, m0 = (lane >> 4) << 3;
#pragma unroll
  for (int nt = 0; nt < 4; ++nt)
#pragma unroll
    for (int r = 0; r < 8; ++r)
      E[(size_t)(r0 + m0 + r) * HID + nt * 16 + col] = acc[nt][r];
}

// ctx[b,:] = softmax_n( v . tanh(h@Wfc^T + E[b,n,:]) ) applied to S[b,n,:]
// 256 threads = 8 waves; 2 waves per batch row (split over N). grid B/4.
__global__ void k_attn(const float* __restrict__ hstate, const float* __restrict__ Wfc,
                       const float* __restrict__ vv, const float* __restrict__ E,
                       const float* __restrict__ S, float* __restrict__ ctx_out) {
  __shared__ float hwbuf[4][HID];
  __shared__ float abuf[4][N_SEQ];
  __shared__ float redbuf[8];
  __shared__ float cpart[8][HID];
  int w = threadIdx.x >> 5, lane = threadIdx.x & 31;
  int bi = w >> 1, half = w & 1;
  int b = blockIdx.x * 4 + bi;

  // hW[h'] = sum_h h[b,h]*Wfc[h',h]; this wave computes h' = half*32+lane (f32 VALU, b128 loads)
  {
    const float4* hs4 = (const float4*)(hstate + (size_t)b * HID);
    const float4* Wr4 = (const float4*)(Wfc + (size_t)(half * 32 + lane) * HID);
    float hw = 0.f;
#pragma unroll
    for (int q = 0; q < 16; ++q) {
      float4 a = hs4[q], bq = Wr4[q];
      hw = fmaf(a.x, bq.x, hw); hw = fmaf(a.y, bq.y, hw);
      hw = fmaf(a.z, bq.z, hw); hw = fmaf(a.w, bq.w, hw);
    }
    hwbuf[bi][half * 32 + lane] = hw;
  }
  __syncthreads();

  // scores: this wave owns 4 sequence positions per lane (128 of 256 rows)
  float sc[4];
#pragma unroll
  for (int j = 0; j < 4; ++j) {
    int nn = half * 128 + j * 32 + lane;
    const float4* e4 = (const float4*)(E + ((size_t)b * N_SEQ + nn) * HID);
    float s = 0.f;
#pragma unroll 4
    for (int q = 0; q < 16; ++q) {
      float4 ev = e4[q];
      float4 hv = *(const float4*)&hwbuf[bi][4 * q];
      s = fmaf(vv[4 * q + 0], fast_tanh(hv.x + ev.x), s);
      s = fmaf(vv[4 * q + 1], fast_tanh(hv.y + ev.y), s);
      s = fmaf(vv[4 * q + 2], fast_tanh(hv.z + ev.z), s);
      s = fmaf(vv[4 * q + 3], fast_tanh(hv.w + ev.w), s);
    }
    sc[j] = s;
  }
  // split softmax: wave-local max -> pair max via LDS
  float m = fmaxf(fmaxf(sc[0], sc[1]), fmaxf(sc[2], sc[3]));
#pragma unroll
  for (int off = 16; off; off >>= 1) m = fmaxf(m, __shfl_xor(m, off, 32));
  if (lane == 0) redbuf[w] = m;
  __syncthreads();
  m = fmaxf(redbuf[bi * 2], redbuf[bi * 2 + 1]);
  float p[4], sum = 0.f;
#pragma unroll
  for (int j = 0; j < 4; ++j) { p[j] = fast_exp(sc[j] - m); sum += p[j]; }
#pragma unroll
  for (int off = 16; off; off >>= 1) sum += __shfl_xor(sum, off, 32);
  __syncthreads();                       // all reads of redbuf(max) done
  if (lane == 0) redbuf[w] = sum;
  __syncthreads();
  float inv = __builtin_amdgcn_rcpf(redbuf[bi * 2] + redbuf[bi * 2 + 1]);
#pragma unroll
  for (int j = 0; j < 4; ++j) abuf[bi][half * 128 + j * 32 + lane] = p[j] * inv;
  __syncthreads();

  // ctx partial: this wave accumulates its 128 rows of S, 2 h-values per lane
  int h0 = lane * 2;
  float c0 = 0.f, c1 = 0.f;
  const float* Sb = S + (size_t)b * N_SEQ * HID;
#pragma unroll 8
  for (int jn = 0; jn < 128; ++jn) {
    int nn = half * 128 + jn;
    float an = abuf[bi][nn];
    if ((jn & 7) == 0 && jn + 16 < 128)
      __builtin_prefetch(Sb + (size_t)(nn + 16) * HID + h0, 0, 1);
    float2 sv = *(const float2*)(Sb + (size_t)nn * HID + h0);
    c0 = fmaf(an, sv.x, c0);
    c1 = fmaf(an, sv.y, c1);
  }
  cpart[w][h0] = c0; cpart[w][h0 + 1] = c1;
  __syncthreads();
  if (half == 0) {
    ctx_out[(size_t)b * HID + h0]     = cpart[w][h0]     + cpart[w + 1][h0];
    ctx_out[(size_t)b * HID + h0 + 1] = cpart[w][h0 + 1] + cpart[w + 1][h0 + 1];
  }
}

// LSTM cell: gates = xinp@W_ih^T + h@W_hh^T + b; in-place h,c update.
// grid B/16, 128 threads = 4 waves; wave w owns gate cols [w*64, w*64+64).
__global__ void k_lstm(const float* __restrict__ xinp, float* __restrict__ h,
                       float* __restrict__ c, const __bf16* __restrict__ Wih,
                       const __bf16* __restrict__ Whh, const float* __restrict__ bih,
                       const float* __restrict__ bhh) {
  __shared__ float gbuf[16][4 * HID];
  int w = threadIdx.x >> 5, lane = threadIdx.x & 31;
  int b0 = blockIdx.x * 16;
  v8f acc[4] = {};
#pragma unroll
  for (int kk = 0; kk < 4; ++kk) {
    v16bf a = load_a_frag_f32(xinp + (size_t)b0 * DIM + kk * 32, DIM);
#pragma unroll
    for (int nt = 0; nt < 4; ++nt)
      acc[nt] = wmma_bf16(a, load_b_frag(Wih + (size_t)(w * 64 + nt * 16) * DIM + kk * 32, DIM), acc[nt]);
  }
#pragma unroll
  for (int kk = 0; kk < 2; ++kk) {
    v16bf a = load_a_frag_f32(h + (size_t)b0 * HID + kk * 32, HID);
#pragma unroll
    for (int nt = 0; nt < 4; ++nt)
      acc[nt] = wmma_bf16(a, load_b_frag(Whh + (size_t)(w * 64 + nt * 16) * HID + kk * 32, HID), acc[nt]);
  }
  int col = lane & 15, m0 = (lane >> 4) << 3;
#pragma unroll
  for (int nt = 0; nt < 4; ++nt)
#pragma unroll
    for (int r = 0; r < 8; ++r)
      gbuf[m0 + r][w * 64 + nt * 16 + col] = acc[nt][r];
  __syncthreads();

  for (int p = threadIdx.x; p < 16 * HID; p += 128) {
    int r = p >> 6, hh = p & 63;
    int b = b0 + r;
    float ig = gbuf[r][hh]           + bih[hh]           + bhh[hh];
    float fg = gbuf[r][HID + hh]     + bih[HID + hh]     + bhh[HID + hh];
    float gg = gbuf[r][2 * HID + hh] + bih[2 * HID + hh] + bhh[2 * HID + hh];
    float og = gbuf[r][3 * HID + hh] + bih[3 * HID + hh] + bhh[3 * HID + hh];
    float cc = c[(size_t)b * HID + hh];
    cc = fast_sig(fg) * cc + fast_sig(ig) * fast_tanh(gg);
    float hv = fast_sig(og) * fast_tanh(cc);
    c[(size_t)b * HID + hh] = cc;
    h[(size_t)b * HID + hh] = hv;
  }
}

// xi = tanh([h, ident_ctx, ctx] @ W_pre^T + b_pre); write ys[t] and next xinp = [xi, ctx].
// grid B/64, 128 threads = 4 waves x 16 rows.
__global__ void k_pre(const float* __restrict__ h, const float* __restrict__ ic,
                      const float* __restrict__ ctx, const __bf16* __restrict__ Wpre,
                      const float* __restrict__ bpre, float* __restrict__ ys_t,
                      float* __restrict__ xinp) {
  int w = threadIdx.x >> 5, lane = threadIdx.x & 31;
  int b0 = blockIdx.x * 64 + w * 16;
  const float* srcs[3] = { h + (size_t)b0 * HID, ic + (size_t)b0 * HID, ctx + (size_t)b0 * HID };
  v8f acc[4] = {};
#pragma unroll
  for (int kk = 0; kk < 6; ++kk) {
    v16bf a = load_a_frag_f32(srcs[kk >> 1] + (kk & 1) * 32, HID);
#pragma unroll
    for (int nt = 0; nt < 4; ++nt)
      acc[nt] = wmma_bf16(a, load_b_frag(Wpre + (size_t)(nt * 16) * (3 * HID) + kk * 32, 3 * HID), acc[nt]);
  }
  int col = lane & 15, m0 = (lane >> 4) << 3;
#pragma unroll
  for (int nt = 0; nt < 4; ++nt) {
    int hc = nt * 16 + col;
    float bi = bpre[hc];
#pragma unroll
    for (int r = 0; r < 8; ++r) {
      int b = b0 + m0 + r;
      float xi = fast_tanh(acc[nt][r] + bi);
      ys_t[(size_t)b * HID + hc] = xi;
      xinp[(size_t)b * DIM + hc] = xi;
    }
  }
  for (int idx = threadIdx.x; idx < 64 * HID; idx += 128) {
    int br = idx >> 6, cc = idx & 63;
    int b = blockIdx.x * 64 + br;
    xinp[(size_t)b * DIM + HID + cc] = ctx[(size_t)b * HID + cc];
  }
}

// out[b,t,o] = ys[t,b,:] . W_cal[o,:] + b_cal[o]
__global__ void k_out(const float* __restrict__ ys, const float* __restrict__ Wcal,
                      const float* __restrict__ bcal, float* __restrict__ out) {
  int id = blockIdx.x * blockDim.x + threadIdx.x;
  if (id >= BATCH * TSTEPS) return;
  int b = id / TSTEPS, t = id % TSTEPS;
  const float4* y4 = (const float4*)(ys + ((size_t)t * BATCH + b) * HID);
#pragma unroll
  for (int o = 0; o < ODIM; ++o) {
    const float4* w4 = (const float4*)(Wcal + o * HID);
    float s = bcal[o];
#pragma unroll
    for (int q = 0; q < 16; ++q) {
      float4 a = y4[q], bq = w4[q];
      s = fmaf(a.x, bq.x, s); s = fmaf(a.y, bq.y, s);
      s = fmaf(a.z, bq.z, s); s = fmaf(a.w, bq.w, s);
    }
    out[((size_t)b * TSTEPS + t) * ODIM + o] = s;
  }
}

__global__ void k_init(const float* __restrict__ i0, const float* __restrict__ x,
                       float* __restrict__ h, float* __restrict__ c,
                       float* __restrict__ xinp) {
  int idx = blockIdx.x * blockDim.x + threadIdx.x;
  if (idx < BATCH * DIM) xinp[idx] = x[(size_t)(N_SEQ - 1) * BATCH * DIM + idx];
  if (idx < BATCH * HID) { float t = i0[idx]; h[idx] = t; c[idx] = t; }
}

// ---------------------------------------------------------------- launcher
extern "C" void kernel_launch(void* const* d_in, const int* in_sizes, int n_in,
                              void* d_out, int out_size, void* d_ws, size_t ws_size,
                              hipStream_t stream) {
  const float* x     = (const float*)d_in[0];
  const float* i0    = (const float*)d_in[1];
  const float* W_id  = (const float*)d_in[2];
  const float* b_id  = (const float*)d_in[3];
  const float* W_xl  = (const float*)d_in[4];
  const float* b_xl  = (const float*)d_in[5];
  const float* Wfc_i = (const float*)d_in[6];
  const float* v_i   = (const float*)d_in[7];
  const float* Wfc   = (const float*)d_in[8];
  const float* v     = (const float*)d_in[9];
  const float* W_ih  = (const float*)d_in[10];
  const float* W_hh  = (const float*)d_in[11];
  const float* b_ih  = (const float*)d_in[12];
  const float* b_hh  = (const float*)d_in[13];
  const float* W_pre = (const float*)d_in[14];
  const float* b_pre = (const float*)d_in[15];
  const float* W_cal = (const float*)d_in[16];
  const float* b_cal = (const float*)d_in[17];
  float* out = (float*)d_out;

  char* base = (char*)d_ws;
  size_t off = 0;
  auto carve = [&](size_t bytes) -> void* {
    void* p = base + off;
    off = (off + bytes + 255) & ~(size_t)255;
    return p;
  };
  const size_t RNH = (size_t)BATCH * N_SEQ * HID;
  __bf16* W_idb  = (__bf16*)carve((size_t)HID * DIM * 2);
  __bf16* W_xlb  = (__bf16*)carve((size_t)HID * DIM * 2);
  __bf16* Wfc_ib = (__bf16*)carve((size_t)HID * HID * 2);
  __bf16* Wfcb   = (__bf16*)carve((size_t)HID * HID * 2);
  __bf16* W_ihb  = (__bf16*)carve((size_t)4 * HID * DIM * 2);
  __bf16* W_hhb  = (__bf16*)carve((size_t)4 * HID * HID * 2);
  __bf16* W_preb = (__bf16*)carve((size_t)HID * 3 * HID * 2);
  float* x_ident = (float*)carve(RNH * 4);
  float* x_lat   = (float*)carve(RNH * 4);
  float* encA    = (float*)carve(RNH * 4);
  float* encB    = (float*)carve(RNH * 4);
  float* hbuf    = (float*)carve((size_t)BATCH * HID * 4);
  float* cbuf    = (float*)carve((size_t)BATCH * HID * 4);
  float* xinp    = (float*)carve((size_t)BATCH * DIM * 4);
  float* ictx    = (float*)carve((size_t)BATCH * HID * 4);
  float* ctx     = (float*)carve((size_t)BATCH * HID * 4);
  float* ys      = (float*)carve((size_t)TSTEPS * BATCH * HID * 4);

  auto cvt = [&](const float* s, __bf16* d, int n) {
    k_cvt<<<(n + 255) / 256, 256, 0, stream>>>(s, d, n);
  };
  cvt(W_id, W_idb, HID * DIM);
  cvt(W_xl, W_xlb, HID * DIM);
  cvt(Wfc_i, Wfc_ib, HID * HID);
  cvt(Wfc, Wfcb, HID * HID);
  cvt(W_ih, W_ihb, 4 * HID * DIM);
  cvt(W_hh, W_hhb, 4 * HID * HID);
  cvt(W_pre, W_preb, HID * 3 * HID);

  // fused dual projection (x read once) + loop-invariant attention pre-GEMMs
  dim3 gx(BATCH / 64, N_SEQ);
  k_gemm_xw2<<<gx, 128, 0, stream>>>(x, W_idb, b_id, x_ident, W_xlb, b_xl, x_lat);
  int Rblk = (BATCH * N_SEQ) / 64;
  k_gemm_enc<<<Rblk, 128, 0, stream>>>(x_ident, Wfc_ib, encA);
  k_gemm_enc<<<Rblk, 128, 0, stream>>>(x_lat, Wfcb, encB);

  k_init<<<(BATCH * DIM + 255) / 256, 256, 0, stream>>>(i0, x, hbuf, cbuf, xinp);
  k_attn<<<BATCH / 4, 256, 0, stream>>>(i0, Wfc_i, v_i, encA, x_ident, ictx);

  for (int t = 0; t < TSTEPS; ++t) {
    k_lstm<<<BATCH / 16, 128, 0, stream>>>(xinp, hbuf, cbuf, W_ihb, W_hhb, b_ih, b_hh);
    k_attn<<<BATCH / 4, 256, 0, stream>>>(hbuf, Wfc, v, encB, x_lat, ctx);
    k_pre<<<BATCH / 64, 128, 0, stream>>>(hbuf, ictx, ctx, W_preb, b_pre,
                                          ys + (size_t)t * BATCH * HID, xinp);
  }
  k_out<<<(BATCH * TSTEPS + 255) / 256, 256, 0, stream>>>(ys, W_cal, b_cal, out);
  (void)in_sizes; (void)n_in; (void)out_size; (void)ws_size;
}
